// MiniMaxM2DecoderLayer_36017595744844
// MI455X (gfx1250) — compile-verified, matrix operand order checked by reading, not run on hardware
//
#include <hip/hip_runtime.h>
#include <hip/hip_bf16.h>

typedef __attribute__((ext_vector_type(16))) __bf16 v16bf;
typedef __attribute__((ext_vector_type(8)))  float  v8f;

#define HID   3072
#define NH    24
#define NKV   8
#define HD    128
#define NE    8
#define FF    1536
#define TT    1024
#define QKV_N ((NH + 2 * NKV) * HD)   // 5120
#define KDOWN (NE * FF)               // 12288
#define LDSS  36                      // padded LDS row stride for 32-wide tiles (bf16)
#define KVS   132                     // padded LDS row stride for 128-wide K/V tiles

// ---------------------------------------------------------------- WMMA helpers

static __device__ __forceinline__ v8f wmma_bf16(v16bf a, v16bf b, v8f c) {
  // D = A(16x32 bf16) * B(32x16 bf16) + C(16x16 f32)
  return __builtin_amdgcn_wmma_f32_16x16x32_bf16(false, a, false, b, (short)0, c,
                                                 false, false);
}

static __device__ __forceinline__ v8f zero8() {
  v8f z;
#pragma unroll
  for (int i = 0; i < 8; ++i) z[i] = 0.0f;
  return z;
}

// A/B fragment from a row-major LDS tile: lane&15 -> row (M or N),
// lane[4] selects K half per CDNA5 16-bit layout, kofs = K window start.
static __device__ __forceinline__ v16bf frag_rowmajor(const __bf16* base, int stride,
                                                      int lane, int kofs) {
  int r  = lane & 15;
  int kb = ((lane >> 4) << 3) + kofs;
  const __bf16* p = base + r * stride + kb;
  v16bf f;
#pragma unroll
  for (int i = 0; i < 8; ++i) f[i] = p[i];          // K = kb..kb+7
#pragma unroll
  for (int i = 0; i < 8; ++i) f[8 + i] = p[16 + i]; // K = kb+16..kb+23
  return f;
}

// B fragment where the matmul-K runs DOWN the LDS rows and N is a fixed column.
static __device__ __forceinline__ v16bf frag_colgather(const __bf16* base, int stride,
                                                       int lane, int col) {
  int kb = (lane >> 4) << 3;
  v16bf f;
#pragma unroll
  for (int i = 0; i < 8; ++i) f[i] = base[(kb + i) * stride + col];
#pragma unroll
  for (int i = 0; i < 8; ++i) f[8 + i] = base[(16 + kb + i) * stride + col];
  return f;
}

// ---------------------------------------------------------------- tile loaders
// f32 global -> bf16 LDS, 32-wide K tiles.

static __device__ __forceinline__ void load_tile_128x32(const float* src, int ld,
                                                        __bf16* dst, int tid) {
  int r = tid >> 1;            // 2 threads per row, 128 rows
  int c = (tid & 1) << 4;      // 16 floats each
  const float* s = src + (size_t)r * ld + c;
  __bf16* d = dst + r * LDSS + c;
  __builtin_prefetch(s + 32, 0, 3);   // next K tile -> global_prefetch_b8
#pragma unroll
  for (int i = 0; i < 16; i += 4) {
    float4 v = *(const float4*)(s + i);
    d[i + 0] = (__bf16)v.x; d[i + 1] = (__bf16)v.y;
    d[i + 2] = (__bf16)v.z; d[i + 3] = (__bf16)v.w;
  }
}

static __device__ __forceinline__ void load_tile_64x32(const float* src, int ld,
                                                       __bf16* dst, int tid) {
  int r = tid >> 2;            // 4 threads per row, 64 rows
  int c = (tid & 3) << 3;      // 8 floats each
  const float* s = src + (size_t)r * ld + c;
  __bf16* d = dst + r * LDSS + c;
  __builtin_prefetch(s + 32, 0, 3);
#pragma unroll
  for (int i = 0; i < 8; i += 4) {
    float4 v = *(const float4*)(s + i);
    d[i + 0] = (__bf16)v.x; d[i + 1] = (__bf16)v.y;
    d[i + 2] = (__bf16)v.z; d[i + 3] = (__bf16)v.w;
  }
}

// ---------------------------------------------------------------- RMSNorm

__global__ void __launch_bounds__(256)
rmsnorm_kernel(const float* __restrict__ x, const float* __restrict__ w,
               float* __restrict__ out, int W) {
  __shared__ float red[256];
  int t = blockIdx.x, tid = threadIdx.x;
  const float* xp = x + (size_t)t * W;
  float s = 0.0f;
  for (int i = tid; i < W; i += 256) { float v = xp[i]; s += v * v; }
  red[tid] = s; __syncthreads();
  for (int o = 128; o > 0; o >>= 1) { if (tid < o) red[tid] += red[tid + o]; __syncthreads(); }
  float inv = rsqrtf(red[0] / (float)W + 1e-6f);
  float* op = out + (size_t)t * W;
  for (int i = tid; i < W; i += 256) op[i] = xp[i] * inv * w[i];
}

// ---------------------------------------------------------------- generic GEMM
// C[M,N] = A[M,K] @ B[N,K]^T (+ resid). 128x128 block, 8 waves of 32x64.

template <bool ADD>
__global__ void __launch_bounds__(256)
gemm_wmma(const float* __restrict__ A, int lda, const float* __restrict__ B, int ldb,
          const float* __restrict__ resid, float* __restrict__ C, int ldc, int K) {
  __shared__ __bf16 As[128 * LDSS];
  __shared__ __bf16 Bs[128 * LDSS];
  int tid = threadIdx.x, lane = tid & 31, w = tid >> 5;
  int mb = blockIdx.y * 128, nb = blockIdx.x * 128;
  int wm = (w & 3) * 32, wn = (w >> 2) * 64;

  v8f acc[2][4];
#pragma unroll
  for (int mt = 0; mt < 2; ++mt)
#pragma unroll
    for (int nt = 0; nt < 4; ++nt) acc[mt][nt] = zero8();

  for (int k0 = 0; k0 < K; k0 += 32) {
    load_tile_128x32(A + (size_t)mb * lda + k0, lda, As, tid);
    load_tile_128x32(B + (size_t)nb * ldb + k0, ldb, Bs, tid);
    __syncthreads();
    v16bf a[2], b[4];
#pragma unroll
    for (int mt = 0; mt < 2; ++mt) a[mt] = frag_rowmajor(As + (wm + mt * 16) * LDSS, LDSS, lane, 0);
#pragma unroll
    for (int nt = 0; nt < 4; ++nt) b[nt] = frag_rowmajor(Bs + (wn + nt * 16) * LDSS, LDSS, lane, 0);
#pragma unroll
    for (int mt = 0; mt < 2; ++mt)
#pragma unroll
      for (int nt = 0; nt < 4; ++nt) acc[mt][nt] = wmma_bf16(a[mt], b[nt], acc[mt][nt]);
    __syncthreads();
  }

  int hi8 = (lane >> 4) << 3, ln = lane & 15;
#pragma unroll
  for (int mt = 0; mt < 2; ++mt)
#pragma unroll
    for (int nt = 0; nt < 4; ++nt)
#pragma unroll
      for (int v = 0; v < 8; ++v) {
        int row = mb + wm + mt * 16 + hi8 + v;
        int col = nb + wn + nt * 16 + ln;
        float val = acc[mt][nt][v];
        if (ADD) val += resid[(size_t)row * ldc + col];
        C[(size_t)row * ldc + col] = val;
      }
}

// ---------------------------------------------------------------- QK-norm + partial RoPE

__global__ void __launch_bounds__(256)
qknorm_rope_kernel(const float* __restrict__ qkv, const int* __restrict__ positions,
                   const float* __restrict__ qw, const float* __restrict__ kw,
                   float* __restrict__ qs, float* __restrict__ ks, float* __restrict__ vs) {
  __shared__ float red[256];
  int t = blockIdx.x, tid = threadIdx.x;
  const float* p = qkv + (size_t)t * QKV_N;
  float pos = (float)positions[t];

  // rms over q (3072) and k (1024)
  float sq = 0.0f, sk = 0.0f;
  for (int i = tid; i < NH * HD; i += 256) { float v = p[i]; sq += v * v; }
  for (int i = tid; i < NKV * HD; i += 256) { float v = p[NH * HD + i]; sk += v * v; }
  red[tid] = sq; __syncthreads();
  for (int o = 128; o > 0; o >>= 1) { if (tid < o) red[tid] += red[tid + o]; __syncthreads(); }
  float invq = rsqrtf(red[0] / (float)(NH * HD) + 1e-6f);
  __syncthreads();
  red[tid] = sk; __syncthreads();
  for (int o = 128; o > 0; o >>= 1) { if (tid < o) red[tid] += red[tid + o]; __syncthreads(); }
  float invk = rsqrtf(red[0] / (float)(NKV * HD) + 1e-6f);

  // q: norm + rope on first 64 dims of each head
  for (int i = tid; i < NH * HD; i += 256) {
    int hh = i >> 7, d = i & 127;
    float o;
    if (d < 64) {
      int j = d & 31;
      float f = pos * __powf(10000.0f, -(float)j / 32.0f);
      float cs = __cosf(f), sn = __sinf(f);
      int b = hh * HD;
      float x1 = p[b + j] * invq * qw[b + j];
      float x2 = p[b + j + 32] * invq * qw[b + j + 32];
      o = (d < 32) ? (x1 * cs - x2 * sn) : (x2 * cs + x1 * sn);
    } else {
      o = p[i] * invq * qw[i];
    }
    qs[(size_t)t * (NH * HD) + i] = o;
  }
  // k: norm + rope
  const float* pk = p + NH * HD;
  for (int i = tid; i < NKV * HD; i += 256) {
    int hh = i >> 7, d = i & 127;
    float o;
    if (d < 64) {
      int j = d & 31;
      float f = pos * __powf(10000.0f, -(float)j / 32.0f);
      float cs = __cosf(f), sn = __sinf(f);
      int b = hh * HD;
      float x1 = pk[b + j] * invk * kw[b + j];
      float x2 = pk[b + j + 32] * invk * kw[b + j + 32];
      o = (d < 32) ? (x1 * cs - x2 * sn) : (x2 * cs + x1 * sn);
    } else {
      o = pk[i] * invk * kw[i];
    }
    ks[(size_t)t * (NKV * HD) + i] = o;
  }
  // v passthrough
  const float* pv = p + (NH + NKV) * HD;
  for (int i = tid; i < NKV * HD; i += 256)
    vs[(size_t)t * (NKV * HD) + i] = pv[i];
}

// ---------------------------------------------------------------- flash attention (WMMA)
// grid (NH, TT/64), 4 waves; each wave owns 16 query rows.

__global__ void __launch_bounds__(128)
attn_kernel(const float* __restrict__ Q, const float* __restrict__ K,
            const float* __restrict__ V, float* __restrict__ O) {
  __shared__ __bf16 Kt[32 * KVS];
  __shared__ __bf16 Vt[32 * KVS];
  __shared__ __bf16 Pt[4][16 * LDSS];
  int tid = threadIdx.x, lane = tid & 31, w = tid >> 5;
  int h = blockIdx.x;
  int qb0 = blockIdx.y * 64;
  int kvh = h / (NH / NKV);
  int qrow0 = qb0 + w * 16;
  int hi8 = (lane >> 4) << 3, ln = lane & 15;
  const float scale = 0.08838834764831845f;  // 128^-0.5

  // Q fragments (4 chunks of K=32 over HD=128), loaded straight into frag layout
  v16bf aq[4];
  {
    const float* qp = Q + (size_t)(qrow0 + ln) * (NH * HD) + h * HD;
#pragma unroll
    for (int dk = 0; dk < 4; ++dk) {
      v16bf f;
#pragma unroll
      for (int i = 0; i < 8; ++i) f[i] = (__bf16)qp[dk * 32 + hi8 + i];
#pragma unroll
      for (int i = 0; i < 8; ++i) f[8 + i] = (__bf16)qp[dk * 32 + 16 + hi8 + i];
      aq[dk] = f;
    }
  }

  v8f oacc[8];
#pragma unroll
  for (int dt = 0; dt < 8; ++dt) oacc[dt] = zero8();
  float mrun[8], lrun[8];
#pragma unroll
  for (int v = 0; v < 8; ++v) { mrun[v] = -1e30f; lrun[v] = 0.0f; }

  int nkb = qb0 / 32 + 2;  // causal: keys 0..qb0+63
  for (int kb = 0; kb < nkb; ++kb) {
    {  // cooperative K/V tile load (32 x 128), f32 -> bf16
      int r = tid >> 2, c = (tid & 3) << 5;
      const float* ksrc = K + (size_t)(kb * 32 + r) * (NKV * HD) + kvh * HD + c;
      const float* vsrc = V + (size_t)(kb * 32 + r) * (NKV * HD) + kvh * HD + c;
      __bf16* kd = Kt + r * KVS + c;
      __bf16* vd = Vt + r * KVS + c;
#pragma unroll
      for (int i = 0; i < 32; i += 4) {
        float4 kv4 = *(const float4*)(ksrc + i);
        kd[i] = (__bf16)kv4.x; kd[i + 1] = (__bf16)kv4.y;
        kd[i + 2] = (__bf16)kv4.z; kd[i + 3] = (__bf16)kv4.w;
        float4 vv4 = *(const float4*)(vsrc + i);
        vd[i] = (__bf16)vv4.x; vd[i + 1] = (__bf16)vv4.y;
        vd[i + 2] = (__bf16)vv4.z; vd[i + 3] = (__bf16)vv4.w;
      }
    }
    __syncthreads();

    // S = Q * K^T for two 16-key tiles; masked scores kept in registers
    float sc[2][8];
#pragma unroll
    for (int kt = 0; kt < 2; ++kt) {
      v8f s = zero8();
#pragma unroll
      for (int dk = 0; dk < 4; ++dk) {
        v16bf bk = frag_rowmajor(Kt + kt * 16 * KVS, KVS, lane, dk * 32);
        s = wmma_bf16(aq[dk], bk, s);
      }
      int key = kb * 32 + kt * 16 + ln;
#pragma unroll
      for (int v = 0; v < 8; ++v) {
        float sv = s[v] * scale;
        int qr = qrow0 + hi8 + v;
        sc[kt][v] = (key > qr) ? -1e9f : sv;
      }
    }

    // online softmax per row (rows live in 16-lane halves of the wave)
    float rsc[8];
#pragma unroll
    for (int v = 0; v < 8; ++v) {
      float mv = fmaxf(sc[0][v], sc[1][v]);
#pragma unroll
      for (int m = 1; m < 16; m <<= 1) mv = fmaxf(mv, __shfl_xor(mv, m, 32));
      float mn = fmaxf(mrun[v], mv);
      float r = __expf(mrun[v] - mn);
      float p0 = __expf(sc[0][v] - mn);
      float p1 = __expf(sc[1][v] - mn);
      float rs = p0 + p1;
#pragma unroll
      for (int m = 1; m < 16; m <<= 1) rs += __shfl_xor(rs, m, 32);
      lrun[v] = lrun[v] * r + rs;
      mrun[v] = mn;
      rsc[v] = r;
      // transpose P through wave-private LDS (DS ops in-order within a wave)
      Pt[w][(hi8 + v) * LDSS + ln] = (__bf16)p0;
      Pt[w][(hi8 + v) * LDSS + 16 + ln] = (__bf16)p1;
    }

    // rescale running O, then O += P * V
#pragma unroll
    for (int dt = 0; dt < 8; ++dt)
#pragma unroll
      for (int v = 0; v < 8; ++v) oacc[dt][v] *= rsc[v];

    v16bf ap = frag_rowmajor(Pt[w], LDSS, lane, 0);
#pragma unroll
    for (int dt = 0; dt < 8; ++dt) {
      v16bf bv = frag_colgather(Vt, KVS, lane, dt * 16 + ln);
      oacc[dt] = wmma_bf16(ap, bv, oacc[dt]);
    }
    __syncthreads();
  }

#pragma unroll
  for (int dt = 0; dt < 8; ++dt)
#pragma unroll
    for (int v = 0; v < 8; ++v) {
      int qr = qrow0 + hi8 + v;
      O[(size_t)qr * (NH * HD) + h * HD + dt * 16 + ln] = oacc[dt][v] / lrun[v];
    }
}

// ---------------------------------------------------------------- router (sigmoid top-2)

__global__ void __launch_bounds__(256)
router_kernel(const float* __restrict__ h2, const float* __restrict__ gate_w,
              const float* __restrict__ e_bias, float* __restrict__ combine) {
  __shared__ float red[256];
  __shared__ float logit[NE];
  int t = blockIdx.x, tid = threadIdx.x;
  const float* hp = h2 + (size_t)t * HID;
  for (int e = 0; e < NE; ++e) {
    const float* gp = gate_w + (size_t)e * HID;
    float s = 0.0f;
    for (int i = tid; i < HID; i += 256) s += hp[i] * gp[i];
    red[tid] = s; __syncthreads();
    for (int o = 128; o > 0; o >>= 1) { if (tid < o) red[tid] += red[tid + o]; __syncthreads(); }
    if (tid == 0) logit[e] = red[0];
    __syncthreads();
  }
  if (tid == 0) {
    float p[NE];
    for (int e = 0; e < NE; ++e) p[e] = 1.0f / (1.0f + __expf(-logit[e]));
    int i0 = 0; float b0 = -1e30f;
    for (int e = 0; e < NE; ++e) { float s = p[e] + e_bias[e]; if (s > b0) { b0 = s; i0 = e; } }
    int i1 = 0; float b1 = -1e30f;
    for (int e = 0; e < NE; ++e) {
      if (e == i0) continue;
      float s = p[e] + e_bias[e]; if (s > b1) { b1 = s; i1 = e; }
    }
    float w0 = p[i0], w1 = p[i1], inv = 1.0f / (w0 + w1);
    float* cp = combine + (size_t)t * NE;
    for (int e = 0; e < NE; ++e) cp[e] = 0.0f;
    cp[i0] = w0 * inv; cp[i1] = w1 * inv;
  }
}

// ---------------------------------------------------------------- MoE gate/up fused
// grid (FF/64, TT/128, NE): per block 128 tokens x 64 FF cols, both gate and up GEMMs.

__global__ void __launch_bounds__(256)
moe_gateup_kernel(const float* __restrict__ H2, const float* __restrict__ Wg,
                  const float* __restrict__ Wu, const float* __restrict__ combine,
                  float* __restrict__ act) {
  __shared__ __bf16 As[128 * LDSS];
  __shared__ __bf16 Bg[64 * LDSS];
  __shared__ __bf16 Bu[64 * LDSS];
  int tid = threadIdx.x, lane = tid & 31, w = tid >> 5;
  int e = blockIdx.z;
  int mb = blockIdx.y * 128, nb = blockIdx.x * 64;
  const float* WgE = Wg + (size_t)e * FF * HID;
  const float* WuE = Wu + (size_t)e * FF * HID;

  v8f ag[4], au[4];
#pragma unroll
  for (int nt = 0; nt < 4; ++nt) { ag[nt] = zero8(); au[nt] = zero8(); }

  for (int k0 = 0; k0 < HID; k0 += 32) {
    load_tile_128x32(H2 + (size_t)mb * HID + k0, HID, As, tid);
    load_tile_64x32(WgE + (size_t)nb * HID + k0, HID, Bg, tid);
    load_tile_64x32(WuE + (size_t)nb * HID + k0, HID, Bu, tid);
    __syncthreads();
    v16bf a = frag_rowmajor(As + w * 16 * LDSS, LDSS, lane, 0);
#pragma unroll
    for (int nt = 0; nt < 4; ++nt) {
      v16bf bg = frag_rowmajor(Bg + nt * 16 * LDSS, LDSS, lane, 0);
      v16bf bu = frag_rowmajor(Bu + nt * 16 * LDSS, LDSS, lane, 0);
      ag[nt] = wmma_bf16(a, bg, ag[nt]);
      au[nt] = wmma_bf16(a, bu, au[nt]);
    }
    __syncthreads();
  }

  int hi8 = (lane >> 4) << 3, ln = lane & 15;
#pragma unroll
  for (int nt = 0; nt < 4; ++nt)
#pragma unroll
    for (int v = 0; v < 8; ++v) {
      int t = mb + w * 16 + hi8 + v;
      int col = nb + nt * 16 + ln;
      float g = ag[nt][v], u = au[nt][v];
      float cw = combine[(size_t)t * NE + e];
      float s = g / (1.0f + __expf(-g));        // silu
      act[(size_t)t * KDOWN + e * FF + col] = s * u * cw;
    }
}

// ---------------------------------------------------------------- MoE down GEMM (+resid)
// out[t,n] = resid[t,n] + sum_e sum_f act[t, e*FF+f] * w_down[e][n][f]; K = 12288.

__global__ void __launch_bounds__(256)
moe_down_kernel(const float* __restrict__ act, const float* __restrict__ Wd,
                const float* __restrict__ resid, float* __restrict__ out) {
  __shared__ __bf16 As[128 * LDSS];
  __shared__ __bf16 Bs[128 * LDSS];
  int tid = threadIdx.x, lane = tid & 31, w = tid >> 5;
  int mb = blockIdx.y * 128, nb = blockIdx.x * 128;
  int wm = (w & 3) * 32, wn = (w >> 2) * 64;

  v8f acc[2][4];
#pragma unroll
  for (int mt = 0; mt < 2; ++mt)
#pragma unroll
    for (int nt = 0; nt < 4; ++nt) acc[mt][nt] = zero8();

  for (int k0 = 0; k0 < KDOWN; k0 += 32) {
    int e = k0 / FF;              // FF % 32 == 0 so a K tile never crosses experts
    int f0 = k0 - e * FF;
    load_tile_128x32(act + (size_t)mb * KDOWN + k0, KDOWN, As, tid);
    {  // B tile: rows = HID columns of this expert's w_down
      int r = tid >> 1, c = (tid & 1) << 4;
      const float* s = Wd + ((size_t)e * HID + nb + r) * FF + f0 + c;
      __bf16* d = Bs + r * LDSS + c;
      __builtin_prefetch(s + 32, 0, 3);
#pragma unroll
      for (int i = 0; i < 16; i += 4) {
        float4 v = *(const float4*)(s + i);
        d[i] = (__bf16)v.x; d[i + 1] = (__bf16)v.y;
        d[i + 2] = (__bf16)v.z; d[i + 3] = (__bf16)v.w;
      }
    }
    __syncthreads();
    v16bf a[2], b[4];
#pragma unroll
    for (int mt = 0; mt < 2; ++mt) a[mt] = frag_rowmajor(As + (wm + mt * 16) * LDSS, LDSS, lane, 0);
#pragma unroll
    for (int nt = 0; nt < 4; ++nt) b[nt] = frag_rowmajor(Bs + (wn + nt * 16) * LDSS, LDSS, lane, 0);
#pragma unroll
    for (int mt = 0; mt < 2; ++mt)
#pragma unroll
      for (int nt = 0; nt < 4; ++nt) acc[mt][nt] = wmma_bf16(a[mt], b[nt], acc[mt][nt]);
    __syncthreads();
  }

  int hi8 = (lane >> 4) << 3, ln = lane & 15;
#pragma unroll
  for (int mt = 0; mt < 2; ++mt)
#pragma unroll
    for (int nt = 0; nt < 4; ++nt)
#pragma unroll
      for (int v = 0; v < 8; ++v) {
        int row = mb + wm + mt * 16 + hi8 + v;
        int col = nb + wn + nt * 16 + ln;
        out[(size_t)row * HID + col] =
            resid[(size_t)row * HID + col] + acc[mt][nt][v];
      }
}

// ---------------------------------------------------------------- launch

extern "C" void kernel_launch(void* const* d_in, const int* in_sizes, int n_in,
                              void* d_out, int out_size, void* d_ws, size_t ws_size,
                              hipStream_t stream) {
  const float* hidden     = (const float*)d_in[0];
  const int*   positions  = (const int*)d_in[1];
  const float* w_qkv      = (const float*)d_in[2];
  const float* q_norm_w   = (const float*)d_in[3];
  const float* k_norm_w   = (const float*)d_in[4];
  const float* w_o        = (const float*)d_in[5];
  const float* input_ln_w = (const float*)d_in[6];
  const float* post_ln_w  = (const float*)d_in[7];
  const float* gate_w     = (const float*)d_in[8];
  const float* e_bias     = (const float*)d_in[9];
  const float* w_gate     = (const float*)d_in[10];
  const float* w_up       = (const float*)d_in[11];
  const float* w_down     = (const float*)d_in[12];
  float* out = (float*)d_out;

  // workspace carve (floats)
  float* ws = (float*)d_ws;
  float* h       = ws;  ws += (size_t)TT * HID;
  float* qkv     = ws;  ws += (size_t)TT * QKV_N;
  float* qs      = ws;  ws += (size_t)TT * NH * HD;
  float* ks      = ws;  ws += (size_t)TT * NKV * HD;
  float* vs      = ws;  ws += (size_t)TT * NKV * HD;
  float* attn_o  = ws;  ws += (size_t)TT * NH * HD;
  float* x2      = ws;  ws += (size_t)TT * HID;
  float* h2      = ws;  ws += (size_t)TT * HID;
  float* combine = ws;  ws += (size_t)TT * NE;
  float* actbuf  = ws;  ws += (size_t)TT * KDOWN;

  // 1. input RMSNorm
  rmsnorm_kernel<<<TT, 256, 0, stream>>>(hidden, input_ln_w, h, HID);
  // 2. QKV projection
  gemm_wmma<false><<<dim3(QKV_N / 128, TT / 128), 256, 0, stream>>>(
      h, HID, w_qkv, HID, nullptr, qkv, QKV_N, HID);
  // 3. qk-norm + partial RoPE + v split
  qknorm_rope_kernel<<<TT, 256, 0, stream>>>(qkv, positions, q_norm_w, k_norm_w,
                                             qs, ks, vs);
  // 4. causal GQA flash attention
  attn_kernel<<<dim3(NH, TT / 64), 128, 0, stream>>>(qs, ks, vs, attn_o);
  // 5. O projection + residual
  gemm_wmma<true><<<dim3(HID / 128, TT / 128), 256, 0, stream>>>(
      attn_o, NH * HD, w_o, NH * HD, hidden, x2, HID, NH * HD);
  // 6. post RMSNorm
  rmsnorm_kernel<<<TT, 256, 0, stream>>>(x2, post_ln_w, h2, HID);
  // 7. router: sigmoid + biased top-2 -> combine weights
  router_kernel<<<TT, 256, 0, stream>>>(h2, gate_w, e_bias, combine);
  // 8. MoE gate/up fused + silu*up*combine
  moe_gateup_kernel<<<dim3(FF / 64, TT / 128, NE), 256, 0, stream>>>(
      h2, w_gate, w_up, combine, actbuf);
  // 9. MoE down + residual -> out
  moe_down_kernel<<<dim3(HID / 128, TT / 128), 256, 0, stream>>>(
      actbuf, w_down, x2, out);
}
